// Retriever_24026047054434
// MI455X (gfx1250) — matrix operand order repaired and checked by reading, not compiled, not run on hardware
//
#include <hip/hip_runtime.h>
#include <stdint.h>

// Problem constants (fixed by the reference setup)
#define NQ      16384
#define NC      131072
#define DIM     512
#define TOPK    100
#define NBLK    64        // corpus rows per block iteration (per wave)
#define WAVES   4         // waves per workgroup
#define QTILE   (WAVES * 16)  // 64 query rows per workgroup
#define LISTLEN 128       // padded top-k list stride (100 used)
#define SIMPAD  (NBLK + 2)

typedef __attribute__((ext_vector_type(16))) __bf16 v16bf;
typedef __attribute__((ext_vector_type(8)))  float  v8f;

// Plain-old-data 16-byte chunk (avoids HIP vector-type ctor issues in casts)
typedef struct { uint32_t x, y, z, w; } u32x4;

__device__ __forceinline__ uint16_t f2bf_rne(float f) {
  union { float f; uint32_t u; } cv; cv.f = f;
  uint32_t u = cv.u;
  uint32_t r = u + 0x7FFFu + ((u >> 16) & 1u);   // round-to-nearest-even
  return (uint16_t)(r >> 16);
}

// ---------------------------------------------------------------------------
// Kernel 1: f32 -> bf16 conversion. Makes the corpus 134 MB, i.e. resident in
// the 192 MB global L2 for the compute pass.
// ---------------------------------------------------------------------------
__global__ void cvt_f32_to_bf16(const float* __restrict__ src,
                                uint16_t* __restrict__ dst, int n4) {
  int i      = blockIdx.x * blockDim.x + threadIdx.x;
  int stride = gridDim.x * blockDim.x;
  const float4* s4 = (const float4*)src;
  uint2*        d4 = (uint2*)dst;
  for (; i < n4; i += stride) {
    float4 f = s4[i];
    uint2 o;
    o.x = (uint32_t)f2bf_rne(f.x) | ((uint32_t)f2bf_rne(f.y) << 16);
    o.y = (uint32_t)f2bf_rne(f.z) | ((uint32_t)f2bf_rne(f.w) << 16);
    d4[i] = o;
  }
}

// ---------------------------------------------------------------------------
// Kernel 2: fused bf16-WMMA GEMM + exact streaming top-k.
// Each wave owns 16 query rows end-to-end (A held in 128 VGPRs for the whole
// corpus sweep); 4 waves per WG share the B stream through the WGP$.
// All per-query state is wave-private: no barriers anywhere.
// ---------------------------------------------------------------------------
__global__ __launch_bounds__(128) void topk_wmma(
    const uint16_t* __restrict__ qbf, const uint16_t* __restrict__ cbf,
    float* __restrict__ topv_g, int* __restrict__ topi_g,
    float* __restrict__ outv, int* __restrict__ outi) {

  __shared__ float s_sim[WAVES][16][SIMPAD];  // per-wave 16x64 sim tile
  __shared__ float s_thr[WAVES][16];          // current k-th value per query
  __shared__ int   s_minpos[WAVES][16];       // its slot in the list

  const int lane  = threadIdx.x & 31;
  const int wave  = threadIdx.x >> 5;
  const int half  = lane >> 4;                // 0: lanes 0-15, 1: lanes 16-31
  const int l16   = lane & 15;
  const int qrow0 = blockIdx.x * QTILE + wave * 16;

  // ---- init this wave's 16 top-k lists (global scratch, wave-private) ----
  for (int i = lane; i < 16 * LISTLEN; i += 32) {
    int qi = i >> 7, j = i & (LISTLEN - 1);
    size_t lb = (size_t)(qrow0 + qi) * LISTLEN;
    topv_g[lb + j] = -__builtin_inff();
    topi_g[lb + j] = 0;
  }
  if (lane < 16) { s_thr[wave][lane] = -__builtin_inff(); s_minpos[wave][lane] = 0; }

  // ---- preload A (16 queries x 512 K, bf16) into 128 resident VGPRs ----
  // 16x32 bf16 A layout: lanes 0-15 carry K = {kk..kk+7, kk+16..kk+23} of row
  // l16; lanes 16-31 the +8 shifted chunks.
  const uint16_t* Abase = qbf + (size_t)(qrow0 + l16) * DIM + (half ? 8 : 0);
  v16bf A[16];
#pragma unroll
  for (int s = 0; s < 16; ++s) {
    ((u32x4*)&A[s])[0] = *(const u32x4*)(Abase + s * 32);
    ((u32x4*)&A[s])[1] = *(const u32x4*)(Abase + s * 32 + 16);
  }

  for (int nb = 0; nb < NC; nb += NBLK) {
    if (nb + NBLK < NC) {   // hint next block toward WGP$/L2
      __builtin_prefetch(
          (const void*)(cbf + (size_t)(nb + NBLK + (threadIdx.x & (NBLK - 1))) * DIM), 0, 0);
    }

    // B (32x16 bf16): lane holds column n = corpus row nb+t*16+l16; lanes
    // 0-15 carry K = {kk..kk+7, kk+16..kk+23}, lanes 16-31 the +8 shifted chunks.
    const uint16_t* Bb[4];
#pragma unroll
    for (int t = 0; t < 4; ++t)
      Bb[t] = cbf + (size_t)(nb + t * 16 + l16) * DIM + (half ? 8 : 0);

    v8f acc[4] = {};
#pragma unroll
    for (int s = 0; s < 16; ++s) {
      const int kk = s * 32;
#pragma unroll
      for (int t = 0; t < 4; ++t) {
        v16bf b;
        ((u32x4*)&b)[0] = *(const u32x4*)(Bb[t] + kk);
        ((u32x4*)&b)[1] = *(const u32x4*)(Bb[t] + kk + 16);
        acc[t] = __builtin_amdgcn_wmma_f32_16x16x32_bf16(
            false, A[s], false, b, (short)0, acc[t], false, false);
      }
    }

    // Scatter C tiles to this wave's private LDS region.
    // C/D layout: VGPR r, lane L -> row r+8*(L>=16), col L&15 of the tile.
#pragma unroll
    for (int t = 0; t < 4; ++t) {
#pragma unroll
      for (int r = 0; r < 8; ++r)
        s_sim[wave][r + 8 * half][t * 16 + l16] = acc[t][r];
    }

    // ---- exact top-k update (wave-private; threshold/min-replacement) ----
    for (int qi = 0; qi < 16; ++qi) {
      float thr = s_thr[wave][qi];
      for (;;) {
        // wave-parallel argmax over this block's 64 sims
        float v0 = s_sim[wave][qi][lane];
        float v1 = s_sim[wave][qi][lane + 32];
        float bv; int bn;
        if (v0 >= v1) { bv = v0; bn = lane; } else { bv = v1; bn = lane + 32; }
        for (int off = 16; off > 0; off >>= 1) {
          float ov = __shfl_down(bv, off);
          int   on = __shfl_down(bn, off);
          if (ov > bv || (ov == bv && on < bn)) { bv = ov; bn = on; }
        }
        bv = __shfl(bv, 0); bn = __shfl(bn, 0);
        if (!(bv > thr)) break;             // nothing above the k-th value left
        size_t lb = (size_t)(qrow0 + qi) * LISTLEN;
        if (lane == 0) {
          int mp = s_minpos[wave][qi];
          topv_g[lb + mp] = bv;             // replace current minimum
          topi_g[lb + mp] = nb + bn;
          s_sim[wave][qi][bn] = -__builtin_inff();  // consume candidate
        }
        // recompute threshold = min over the kept K entries (L2-resident)
        float mv = __builtin_inff(); int mp = 0;
        for (int j = lane; j < TOPK; j += 32) {
          float v = topv_g[lb + j];
          if (v < mv) { mv = v; mp = j; }
        }
        for (int off = 16; off > 0; off >>= 1) {
          float ov = __shfl_down(mv, off);
          int   on = __shfl_down(mp, off);
          if (ov < mv) { mv = ov; mp = on; }
        }
        mv = __shfl(mv, 0); mp = __shfl(mp, 0);
        if (lane == 0) { s_thr[wave][qi] = mv; s_minpos[wave][qi] = mp; }
        thr = mv;
      }
    }
  }

  // ---- final: sort each list descending (lax.top_k order) and emit ----
  for (int qi = 0; qi < 16; ++qi) {
    size_t lb = (size_t)(qrow0 + qi) * LISTLEN;
    for (int rank = 0; rank < TOPK; ++rank) {
      float bv = -__builtin_inff(); int bp = LISTLEN;
      for (int j = lane; j < TOPK; j += 32) {
        float v = topv_g[lb + j];
        if (v > bv || (v == bv && j < bp)) { bv = v; bp = j; }
      }
      for (int off = 16; off > 0; off >>= 1) {
        float ov = __shfl_down(bv, off);
        int   op = __shfl_down(bp, off);
        if (ov > bv || (ov == bv && op < bp)) { bv = ov; bp = op; }
      }
      bv = __shfl(bv, 0); bp = __shfl(bp, 0);
      if (lane == 0) {
        size_t o = (size_t)(qrow0 + qi) * TOPK + rank;
        outv[o] = bv;
        outi[o] = topi_g[lb + bp];
        topv_g[lb + bp] = -__builtin_inff();
      }
    }
  }
}

// ---------------------------------------------------------------------------
// Host launcher
// ---------------------------------------------------------------------------
extern "C" void kernel_launch(void* const* d_in, const int* in_sizes, int n_in,
                              void* d_out, int out_size, void* d_ws, size_t ws_size,
                              hipStream_t stream) {
  (void)in_sizes; (void)n_in; (void)out_size; (void)ws_size;

  const float* q = (const float*)d_in[0];   // [NQ, DIM] f32, L2-normalized
  const float* c = (const float*)d_in[1];   // [NC, DIM] f32, L2-normalized
  // d_in[2] = k (100), d_in[3] = chunk_size: fixed by reference, hardcoded.

  // Workspace layout: bf16 query | bf16 corpus | top-k vals | top-k idx
  uint16_t* qbf = (uint16_t*)d_ws;                          // 16.8 MB
  uint16_t* cbf = qbf + (size_t)NQ * DIM;                   // 134 MB (L2-resident)
  float* topv_g = (float*)(cbf + (size_t)NC * DIM);         // 8.4 MB
  int*   topi_g = (int*)(topv_g + (size_t)NQ * LISTLEN);    // 8.4 MB

  cvt_f32_to_bf16<<<2048, 256, 0, stream>>>(q, qbf, (NQ * DIM) / 4);
  cvt_f32_to_bf16<<<8192, 256, 0, stream>>>(c, cbf, (NC * DIM) / 4);

  float* outv = (float*)d_out;                              // [NQ, TOPK] f32
  int*   outi = (int*)d_out + (size_t)NQ * TOPK;            // [NQ, TOPK] i32

  topk_wmma<<<NQ / QTILE, 128, 0, stream>>>(qbf, cbf, topv_g, topi_g, outv, outi);
}
